// memory_unit_75952201662781
// MI455X (gfx1250) — matrix operand order repaired and checked by reading, not compiled, not run on hardware
//
#include <hip/hip_runtime.h>
#include <hip/hip_bf16.h>
#include <math.h>

typedef __attribute__((ext_vector_type(16))) __bf16 v16bf;
typedef __attribute__((ext_vector_type(8)))  float  v8f;

#define BB      32
#define NN      16384
#define MM      64
#define TILES1  (NN / 128)   // kernel1: 8 waves x 16 rows = 128 rows per block
#define BLK2    8            // elementwise kernels: 8 blocks x 2048 elems per batch
#define EPSF    1e-16f
#define COSEPS  1e-8f

// ---------------------------------------------------------------------------
// wave32 XOR-butterfly reductions via ds_swizzle_b32 (group-of-32 mode).
// offset encoding: [14:10]=xor_mask, [9:5]=or_mask, [4:0]=and_mask (ISA §11.2)
// ---------------------------------------------------------------------------
#define SWZ_X1  0x041F
#define SWZ_X2  0x081F
#define SWZ_X4  0x101F
#define SWZ_X8  0x201F
#define SWZ_X16 0x401F

template <int OFF>
__device__ __forceinline__ float swz_xor(float x) {
    return __int_as_float(__builtin_amdgcn_ds_swizzle(__float_as_int(x), OFF));
}

__device__ __forceinline__ float wave_sum32(float v) {
    v += swz_xor<SWZ_X1>(v);
    v += swz_xor<SWZ_X2>(v);
    v += swz_xor<SWZ_X4>(v);
    v += swz_xor<SWZ_X8>(v);
    v += swz_xor<SWZ_X16>(v);
    return v;
}

__device__ __forceinline__ float wave_max32(float v) {
    v = fmaxf(v, swz_xor<SWZ_X1>(v));
    v = fmaxf(v, swz_xor<SWZ_X2>(v));
    v = fmaxf(v, swz_xor<SWZ_X4>(v));
    v = fmaxf(v, swz_xor<SWZ_X8>(v));
    v = fmaxf(v, swz_xor<SWZ_X16>(v));
    return v;
}

// ---------------------------------------------------------------------------
// Kernel 1: content addressing. Each wave computes a 16x16 tile whose columns
// are all dot(memory_row, k) via two v_wmma_f32_16x16x32_bf16 (K=0..31,32..63).
// Row norms accumulated in f32 from the same loaded data. Emits z = beta*cos
// and per-block max partials for the softmax.
// ---------------------------------------------------------------------------
__global__ void __launch_bounds__(256)
k1_content(const float* __restrict__ mem, const float* __restrict__ kvec,
           const float* __restrict__ beta, float* __restrict__ z,
           float* __restrict__ pmax)
{
    const int b    = blockIdx.y;
    const int tid  = threadIdx.x;
    const int wave = tid >> 5;
    const int lane = tid & 31;
    const int n0   = blockIdx.x * 128 + wave * 16;
    const int h    = lane >> 4;      // which K-half of the A layout this lane holds
    const int r    = lane & 15;      // A row held by this lane

    const float* kp = kvec + b * MM;

    // ||k||^2 : 2 elements per lane, full-wave butterfly reduce
    float kv0 = kp[lane];
    float kv1 = kp[lane + 32];
    const float na2 = wave_sum32(kv0 * kv0 + kv1 * kv1);
    const float na  = sqrtf(na2);

    const float* rowp = mem + ((size_t)b * NN + (size_t)(n0 + r)) * MM;

    v8f   acc = {};
    float pn  = 0.f;   // partial ||row||^2 (this lane's K subset)

    #pragma unroll
    for (int c = 0; c < 2; ++c) {
        const int m0 = 32 * c;
        // ---- A tile: elements 0..7 <- K = m0+8h+0..7 ; 8..15 <- K = m0+16+8h+0..7
        float4 f0 = *(const float4*)(rowp + m0 + 8 * h);
        float4 f1 = *(const float4*)(rowp + m0 + 8 * h + 4);
        float4 f2 = *(const float4*)(rowp + m0 + 8 * h + 16);
        float4 f3 = *(const float4*)(rowp + m0 + 8 * h + 20);
        v16bf a;
        a[0]  = (__bf16)f0.x; a[1]  = (__bf16)f0.y; a[2]  = (__bf16)f0.z; a[3]  = (__bf16)f0.w;
        a[4]  = (__bf16)f1.x; a[5]  = (__bf16)f1.y; a[6]  = (__bf16)f1.z; a[7]  = (__bf16)f1.w;
        a[8]  = (__bf16)f2.x; a[9]  = (__bf16)f2.y; a[10] = (__bf16)f2.z; a[11] = (__bf16)f2.w;
        a[12] = (__bf16)f3.x; a[13] = (__bf16)f3.y; a[14] = (__bf16)f3.z; a[15] = (__bf16)f3.w;
        pn += f0.x*f0.x + f0.y*f0.y + f0.z*f0.z + f0.w*f0.w
            + f1.x*f1.x + f1.y*f1.y + f1.z*f1.z + f1.w*f1.w
            + f2.x*f2.x + f2.y*f2.y + f2.z*f2.z + f2.w*f2.w
            + f3.x*f3.x + f3.y*f3.y + f3.z*f3.z + f3.w*f3.w;

        // ---- B tile: all 16 columns = k; elements 0..7 <- K = m0+8h+0..7 ; 8..15 <- K = m0+16+8h+0..7
        const int kb = m0 + 8 * h;
        float4 g0 = *(const float4*)(kp + kb);
        float4 g1 = *(const float4*)(kp + kb + 4);
        float4 g2 = *(const float4*)(kp + kb + 16);
        float4 g3 = *(const float4*)(kp + kb + 20);
        v16bf bm;
        bm[0]  = (__bf16)g0.x; bm[1]  = (__bf16)g0.y; bm[2]  = (__bf16)g0.z; bm[3]  = (__bf16)g0.w;
        bm[4]  = (__bf16)g1.x; bm[5]  = (__bf16)g1.y; bm[6]  = (__bf16)g1.z; bm[7]  = (__bf16)g1.w;
        bm[8]  = (__bf16)g2.x; bm[9]  = (__bf16)g2.y; bm[10] = (__bf16)g2.z; bm[11] = (__bf16)g2.w;
        bm[12] = (__bf16)g3.x; bm[13] = (__bf16)g3.y; bm[14] = (__bf16)g3.z; bm[15] = (__bf16)g3.w;

        acc = __builtin_amdgcn_wmma_f32_16x16x32_bf16(
            /*neg_a=*/false, a, /*neg_b=*/false, bm,
            /*c_mod=*/(short)0, acc, /*reuse_a=*/false, /*reuse_b=*/false);
    }

    // full ||row||^2: lane and lane^16 hold complementary K halves of the same row
    const float nb2 = pn + swz_xor<SWZ_X16>(pn);
    const float nb  = sqrtf(nb2);

    // D layout: lanes 0..7 hold rows 0..7 in acc[lane]; lanes 24..31 hold rows
    // 8..15 in acc[lane-24]; both groups match the row whose norm they hold.
    const bool writer = (lane < 8) || (lane >= 24);
    const int  vi  = (lane < 8) ? lane : (lane - 24);
    const int  row = (lane < 8) ? lane : (lane - 16);
    float dot = acc[0];
    #pragma unroll
    for (int i = 1; i < 8; ++i) dot = (vi == i) ? acc[i] : dot;

    const float cosv = dot / (fmaxf(na, COSEPS) * fmaxf(nb, COSEPS));
    const float zv   = beta[b] * cosv;
    if (writer) z[(size_t)b * NN + n0 + row] = zv;

    // block max for softmax
    const float mz = wave_max32(writer ? zv : -__builtin_inff());
    __shared__ float sm[8];
    if (lane == 0) sm[wave] = mz;
    __syncthreads();
    if (tid == 0) {
        float mm = sm[0];
        #pragma unroll
        for (int i = 1; i < 8; ++i) mm = fmaxf(mm, sm[i]);
        pmax[b * TILES1 + blockIdx.x] = mm;
    }
}

// ---------------------------------------------------------------------------
// Kernel 2: reduce per-block maxima -> per-batch max (deterministic tree)
// ---------------------------------------------------------------------------
__global__ void __launch_bounds__(128)
k2_maxreduce(const float* __restrict__ pmax, float* __restrict__ maxv)
{
    __shared__ float sm[128];
    const int b = blockIdx.x, t = threadIdx.x;
    sm[t] = pmax[b * TILES1 + t];
    __syncthreads();
    for (int s = 64; s > 0; s >>= 1) {
        if (t < s) sm[t] = fmaxf(sm[t], sm[t + s]);
        __syncthreads();
    }
    if (t == 0) maxv[b] = sm[0];
}

// ---------------------------------------------------------------------------
// Kernel 3: e = exp(z - max) (overwrite z) + per-block partial sums
// ---------------------------------------------------------------------------
__global__ void __launch_bounds__(256)
k3_expsum(float* __restrict__ z, const float* __restrict__ maxv,
          float* __restrict__ psumE)
{
    const int b    = blockIdx.y;
    const int base = blockIdx.x * 2048;
    const int t    = threadIdx.x;
    const float mx = maxv[b];
    float* zb = z + (size_t)b * NN;
    float acc = 0.f;
    #pragma unroll
    for (int i = 0; i < 8; ++i) {
        const int j = base + i * 256 + t;
        const float e = expf(zb[j] - mx);
        zb[j] = e;
        acc += e;
    }
    acc = wave_sum32(acc);
    __shared__ float sm[8];
    const int wave = t >> 5, lane = t & 31;
    if (lane == 0) sm[wave] = acc;
    __syncthreads();
    if (t == 0) {
        float s = 0.f;
        #pragma unroll
        for (int i = 0; i < 8; ++i) s += sm[i];
        psumE[b * BLK2 + blockIdx.x] = s;
    }
}

// ---------------------------------------------------------------------------
// Kernel 4: W_c = e/sumE; W_g = g*W_c + (1-g)*W_old (on the fly, including the
// j-1 / j+1 circular neighbors); W_t = s0*Wg(j+1) + s1*Wg(j) + s2*Wg(j-1);
// temp = W_t^gamma -> d_out, plus per-block partial sums of temp.
// ---------------------------------------------------------------------------
__global__ void __launch_bounds__(256)
k4_shiftpow(const float* __restrict__ z, const float* __restrict__ psumE,
            const float* __restrict__ gg, const float* __restrict__ ss,
            const float* __restrict__ gam, const float* __restrict__ wold,
            float* __restrict__ outT, float* __restrict__ psumT)
{
    const int b = blockIdx.y;
    float sumE = 0.f;
    #pragma unroll
    for (int i = 0; i < BLK2; ++i) sumE += psumE[b * BLK2 + i];
    const float inv = 1.f / sumE;
    const float gv = gg[b];
    const float s0 = ss[b * 3 + 0], s1 = ss[b * 3 + 1], s2 = ss[b * 3 + 2];
    const float ga = gam[b];
    const float* zb = z    + (size_t)b * NN;
    const float* wo = wold + (size_t)b * NN;
    float*       ob = outT + (size_t)b * NN;

    const int base = blockIdx.x * 2048;
    const int t    = threadIdx.x;
    float acc = 0.f;
    #pragma unroll
    for (int i = 0; i < 8; ++i) {
        const int j  = base + i * 256 + t;
        const int jm = (j == 0)      ? NN - 1 : j - 1;
        const int jp = (j == NN - 1) ? 0      : j + 1;
        const float wg_p = gv * zb[jp] * inv + (1.f - gv) * wo[jp];
        const float wg_c = gv * zb[j]  * inv + (1.f - gv) * wo[j];
        const float wg_m = gv * zb[jm] * inv + (1.f - gv) * wo[jm];
        const float wt = s0 * wg_p + s1 * wg_c + s2 * wg_m;
        const float tp = powf(wt, ga);
        ob[j] = tp;
        acc += tp;
    }
    acc = wave_sum32(acc);
    __shared__ float sm[8];
    const int wave = t >> 5, lane = t & 31;
    if (lane == 0) sm[wave] = acc;
    __syncthreads();
    if (t == 0) {
        float s = 0.f;
        #pragma unroll
        for (int i = 0; i < 8; ++i) s += sm[i];
        psumT[b * BLK2 + blockIdx.x] = s;
    }
}

// ---------------------------------------------------------------------------
// Kernel 5: W = temp / (sum(temp) + EPS), in place on d_out
// ---------------------------------------------------------------------------
__global__ void __launch_bounds__(256)
k5_normalize(float* __restrict__ out, const float* __restrict__ psumT)
{
    const int b = blockIdx.y;
    float s = 0.f;
    #pragma unroll
    for (int i = 0; i < BLK2; ++i) s += psumT[b * BLK2 + i];
    const float inv = 1.f / (s + EPSF);
    float* ob = out + (size_t)b * NN;
    const int base = blockIdx.x * 2048;
    const int t    = threadIdx.x;
    #pragma unroll
    for (int i = 0; i < 8; ++i) {
        const int j = base + i * 256 + t;
        ob[j] *= inv;
    }
}

extern "C" void kernel_launch(void* const* d_in, const int* in_sizes, int n_in,
                              void* d_out, int out_size, void* d_ws, size_t ws_size,
                              hipStream_t stream)
{
    const float* mem  = (const float*)d_in[0];  // (B, N, M)
    const float* kv   = (const float*)d_in[1];  // (B, M)
    const float* beta = (const float*)d_in[2];  // (B, 1)
    const float* g    = (const float*)d_in[3];  // (B, 1)
    const float* s    = (const float*)d_in[4];  // (B, 3)
    const float* gam  = (const float*)d_in[5];  // (B, 1)
    const float* wold = (const float*)d_in[6];  // (B, N)
    float* out = (float*)d_out;                 // (B, N)

    // workspace carve-up
    float* z     = (float*)d_ws;                // BB*NN
    float* pmax  = z     + (size_t)BB * NN;     // BB*TILES1
    float* maxv  = pmax  + (size_t)BB * TILES1; // BB
    float* psumE = maxv  + BB;                  // BB*BLK2
    float* psumT = psumE + (size_t)BB * BLK2;   // BB*BLK2

    dim3 g1(TILES1, BB);
    k1_content<<<g1, 256, 0, stream>>>(mem, kv, beta, z, pmax);

    k2_maxreduce<<<BB, 128, 0, stream>>>(pmax, maxv);

    dim3 g2(BLK2, BB);
    k3_expsum<<<g2, 256, 0, stream>>>(z, maxv, psumE);
    k4_shiftpow<<<g2, 256, 0, stream>>>(z, psumE, g, s, gam, wold, out, psumT);
    k5_normalize<<<g2, 256, 0, stream>>>(out, psumT);
}
